// CBOWSG_76991583748533
// MI455X (gfx1250) — compile-verified, matrix operand order checked by reading, not run on hardware
//
#include <hip/hip_runtime.h>
#include <hip/hip_bf16.h>

#define B_    256
#define S_    64
#define N_    64
#define EMB_  300
#define HID_  600
#define LDE_  320    // EMB padded to multiple of 32
#define LDH_  608    // HID padded to multiple of 32
#define LDX_  1824   // 6*EMB padded to multiple of 32

typedef __attribute__((ext_vector_type(16))) _Float16 v16h;
typedef __attribute__((ext_vector_type(8)))  _Float16 v8h;
typedef __attribute__((ext_vector_type(8)))  float    v8f;

// ---------------------------------------------------------------------------
// Load one 16x32 (rows x K) f16 WMMA fragment from an f16 row-major matrix
// whose leading dim is a multiple of 32 (branch-free, two 16B loads).
// ISA layout (CDNA5 ISA, 16-bit A 16x32): lanes 0-15 hold row
// M=lane, K={0..7,16..23}; lanes 16-31 hold row M=lane-16, K={8..15,24..31}.
// ---------------------------------------------------------------------------
__device__ __forceinline__ v16h load_frag_h(const _Float16* __restrict__ base,
                                            int row, int ld, int kbase, int lane) {
  const int off = (lane & 16) ? 8 : 0;
  const _Float16* src = base + (size_t)row * ld + kbase + off;
  v8h lo = *(const v8h*)src;          // 16B aligned: ld%8==0, kbase%32==0, off in {0,8}
  v8h hi = *(const v8h*)(src + 16);
  v16h f;
#pragma unroll
  for (int i = 0; i < 8; ++i) { f[i] = lo[i]; f[i + 8] = hi[i]; }
  return f;
}

// ---------------------------------------------------------------------------
// C[m,n] = sum_k A[m,k] * Bw[n,k]  (+bias[n], optional ReLU), all f16 operands,
// f32 accumulate, f16 store. ALL dims are pre-padded multiples of 32 so the
// kernel is completely branch-free. One wave owns a 32x32 C tile (2x2 WMMA
// blocking); 4 waves per block tile along M; grid.z batches weight matrices.
// ---------------------------------------------------------------------------
__global__ void __launch_bounds__(128)
gemm_wmma_f16(const _Float16* __restrict__ A, int lda,
              const _Float16* __restrict__ Bw, int ldb, long long strideB,
              _Float16* __restrict__ C, int ldc, long long strideC,
              int K, const float* __restrict__ bias, int relu) {
  const int lane  = threadIdx.x & 31;
  const int wave  = threadIdx.x >> 5;
  const int r     = lane & 15;
  const int mbase = (blockIdx.x * 4 + wave) * 32;
  const int nbase = blockIdx.y * 32;
  const _Float16* Bp = Bw + (size_t)blockIdx.z * strideB;
  _Float16*       Cp = C  + (size_t)blockIdx.z * strideC;

  v8f acc00 = {}, acc01 = {}, acc10 = {}, acc11 = {};

  for (int k = 0; k < K; k += 32) {
    if (k + 64 <= K) {  // speculative prefetch of next K chunk (global_prefetch_b8)
      __builtin_prefetch(A  + (size_t)(mbase + r) * lda + k + 32, 0, 1);
      __builtin_prefetch(Bp + (size_t)(nbase + r) * ldb + k + 32, 0, 1);
    }
    v16h a0 = load_frag_h(A,  mbase + r,      lda, k, lane);
    v16h a1 = load_frag_h(A,  mbase + 16 + r, lda, k, lane);
    v16h b0 = load_frag_h(Bp, nbase + r,      ldb, k, lane);
    v16h b1 = load_frag_h(Bp, nbase + 16 + r, ldb, k, lane);
    acc00 = __builtin_amdgcn_wmma_f32_16x16x32_f16(false, a0, false, b0, (short)0, acc00, false, false);
    acc01 = __builtin_amdgcn_wmma_f32_16x16x32_f16(false, a0, false, b1, (short)0, acc01, false, false);
    acc10 = __builtin_amdgcn_wmma_f32_16x16x32_f16(false, a1, false, b0, (short)0, acc10, false, false);
    acc11 = __builtin_amdgcn_wmma_f32_16x16x32_f16(false, a1, false, b1, (short)0, acc11, false, false);
  }

  // C/D layout: lane = N (mod 16); VGPR v -> M = v (lanes 0-15) or v+8 (lanes 16-31)
  const int   mofs = (lane & 16) ? 8 : 0;
  const int   nn0  = nbase + r;
  const int   nn1  = nn0 + 16;
  const float bv0  = bias ? bias[nn0] : 0.0f;   // bias pre-padded: no guard
  const float bv1  = bias ? bias[nn1] : 0.0f;
#pragma unroll
  for (int v = 0; v < 8; ++v) {
    const int mm0 = mbase + v + mofs;
    const int mm1 = mm0 + 16;
    float t00 = acc00[v] + bv0, t01 = acc01[v] + bv1;
    float t10 = acc10[v] + bv0, t11 = acc11[v] + bv1;
    if (relu) { t00 = fmaxf(t00, 0.f); t01 = fmaxf(t01, 0.f);
                t10 = fmaxf(t10, 0.f); t11 = fmaxf(t11, 0.f); }
    Cp[(size_t)mm0 * ldc + nn0] = (_Float16)t00;
    Cp[(size_t)mm0 * ldc + nn1] = (_Float16)t01;
    Cp[(size_t)mm1 * ldc + nn0] = (_Float16)t10;
    Cp[(size_t)mm1 * ldc + nn1] = (_Float16)t11;
  }
}

// f32 [nbatch, rows, cols] -> zero-padded f16 [nbatch, drows, dcols]
__global__ void cvt_pad_f16_kernel(const float* __restrict__ src, _Float16* __restrict__ dst,
                                   int rows, int cols, int drows, int dcols, int nbatch) {
  const long long total = (long long)nbatch * drows * dcols;
  for (long long i = (long long)blockIdx.x * blockDim.x + threadIdx.x; i < total;
       i += (long long)gridDim.x * blockDim.x) {
    int c = (int)(i % dcols);
    long long t = i / dcols;
    int rr = (int)(t % drows);
    int z  = (int)(t / drows);
    float v = (rr < rows && c < cols) ? src[((long long)z * rows + rr) * cols + c] : 0.0f;
    dst[i] = (_Float16)v;
  }
}

// f32 [n] -> zero-padded f32 [dn]
__global__ void pad_bias_kernel(const float* __restrict__ src, float* __restrict__ dst,
                                int n, int dn) {
  int i = blockIdx.x * blockDim.x + threadIdx.x;
  if (i < dn) dst[i] = (i < n) ? src[i] : 0.0f;
}

// rep[b,:] = sum_s E_tab[tokens[b,s], :]
__global__ void embed_sum_kernel(const float* __restrict__ E_tab,
                                 const int* __restrict__ tokens,
                                 float* __restrict__ rep) {
  int b = blockIdx.x, e = threadIdx.x;
  if (e >= EMB_) return;
  float s = 0.0f;
  for (int t = 0; t < S_; ++t) {
    int tok = tokens[b * S_ + t];
    s += E_tab[(size_t)tok * EMB_ + e];
  }
  rep[(size_t)b * EMB_ + e] = s;
}

// word[b,n,:]=(idx==-1)?1:E_tab[tokens[b,idx],:]; H16 = word (f16, zero-padded)
__global__ void gather_word_init_kernel(const float* __restrict__ E_tab,
                                        const int* __restrict__ tokens,
                                        const int* __restrict__ word_idx,
                                        float* __restrict__ word,
                                        _Float16* __restrict__ H16) {
  int b = blockIdx.x, n = blockIdx.y, e = threadIdx.x;
  if (e >= LDE_) return;
  float v = 0.0f;
  if (e < EMB_) {
    int wi = word_idx[b * N_ + n];
    if (wi < 0) v = 1.0f;
    else {
      int tok = tokens[b * S_ + wi];
      v = E_tab[(size_t)tok * EMB_ + e];
    }
    word[((size_t)b * N_ + n) * EMB_ + e] = v;
  }
  H16[((size_t)b * N_ + n) * LDE_ + e] = (_Float16)v;  // pads written as 0
}

// Parent p: mean over children at level lvl of (rel in [0,2] ? T16[rel] : H16),
// H16[p] = relu(word[p]*mean) if it has children. Level l-1 writes vs level l
// reads are disjoint -> in-place safe. Pads (e>=300) stay zero.
__global__ void tree_combine_kernel(_Float16* __restrict__ H16,
                                    const float* __restrict__ word,
                                    const _Float16* __restrict__ T16,  // [3, B*N, LDE_]
                                    const int* __restrict__ rel_id,
                                    const int* __restrict__ parent,
                                    const int* __restrict__ level,
                                    int lvl) {
  int b = blockIdx.x, p = blockIdx.y, e = threadIdx.x;
  if (e >= EMB_) return;
  float sum = 0.0f;
  int cnt = 0;
  for (int c = 0; c < N_; ++c) {
    if (parent[c] == p && level[c] == lvl) {
      ++cnt;
      int rl = rel_id[b * N_ + c];
      float t = (rl >= 0 && rl <= 2)
                    ? (float)T16[((size_t)rl * (B_ * N_) + (size_t)b * N_ + c) * LDE_ + e]
                    : (float)H16[((size_t)b * N_ + c) * LDE_ + e];
      sum += t;
    }
  }
  if (cnt > 0) {
    float mean = sum / (float)cnt;
    float w = word[((size_t)b * N_ + p) * EMB_ + e];
    float h = fmaxf(w * mean, 0.0f);
    H16[((size_t)b * N_ + p) * LDE_ + e] = (_Float16)h;
  }
}

// x16 = concat(p_rep, h_rep, p_rep-h_rep, p_rep*h_rep, p_sg-h_sg, p_sg*h_sg), f16 padded
__global__ void build_x_kernel(const float* __restrict__ p_rep,
                               const float* __restrict__ h_rep,
                               const _Float16* __restrict__ Hp16,
                               const _Float16* __restrict__ Hh16,
                               _Float16* __restrict__ x16) {
  int b = blockIdx.x, e = threadIdx.x;
  _Float16* xb = x16 + (size_t)b * LDX_;
  if (e < EMB_) {
    float pr = p_rep[(size_t)b * EMB_ + e];
    float hr = h_rep[(size_t)b * EMB_ + e];
    float ps = (float)Hp16[((size_t)b * N_ + 0) * LDE_ + e];  // root node 0
    float hs = (float)Hh16[((size_t)b * N_ + 0) * LDE_ + e];
    xb[e]            = (_Float16)pr;
    xb[EMB_ + e]     = (_Float16)hr;
    xb[2 * EMB_ + e] = (_Float16)(pr - hr);
    xb[3 * EMB_ + e] = (_Float16)(pr * hr);
    xb[4 * EMB_ + e] = (_Float16)(ps - hs);
    xb[5 * EMB_ + e] = (_Float16)(ps * hs);
  }
  if (e < LDX_ - 6 * EMB_) xb[6 * EMB_ + e] = (_Float16)0.0f;  // K pad
}

// out[b,j] = h3[b,:] . Wout[j,:] + bout[j]   (3 waves per block, shuffle reduce)
__global__ void out_layer_kernel(const _Float16* __restrict__ h3,
                                 const float* __restrict__ Wout,
                                 const float* __restrict__ bout,
                                 float* __restrict__ out) {
  int b = blockIdx.x;
  int j = threadIdx.x >> 5;
  int lane = threadIdx.x & 31;
  float s = 0.0f;
  for (int k = lane; k < HID_; k += 32)
    s += (float)h3[(size_t)b * LDH_ + k] * Wout[(size_t)j * HID_ + k];
  for (int off = 16; off > 0; off >>= 1) s += __shfl_down(s, off, 32);
  if (lane == 0) out[b * 3 + j] = s + bout[j];
}

static inline size_t align256(size_t x) { return (x + 255) & ~(size_t)255; }

extern "C" void kernel_launch(void* const* d_in, const int* in_sizes, int n_in,
                              void* d_out, int out_size, void* d_ws, size_t ws_size,
                              hipStream_t stream) {
  const int*   premise_x    = (const int*)d_in[0];
  const int*   hypothesis_x = (const int*)d_in[1];
  const int*   p_word_idx   = (const int*)d_in[2];
  const int*   p_rel_id     = (const int*)d_in[3];
  const int*   h_word_idx   = (const int*)d_in[4];
  const int*   h_rel_id     = (const int*)d_in[5];
  const int*   parent       = (const int*)d_in[6];
  const int*   level        = (const int*)d_in[7];
  const float* E_tab        = (const float*)d_in[8];
  const float* W_enc        = (const float*)d_in[9];
  const float* W0_w         = (const float*)d_in[10];
  const float* W0_b         = (const float*)d_in[11];
  const float* W1_w         = (const float*)d_in[12];
  const float* W1_b         = (const float*)d_in[13];
  const float* W2_w         = (const float*)d_in[14];
  const float* W2_b         = (const float*)d_in[15];
  const float* Wout_w       = (const float*)d_in[16];
  const float* Wout_b       = (const float*)d_in[17];
  float* out = (float*)d_out;

  const int M_nodes = B_ * N_;  // 16384

  // ---- workspace carve-up ----
  char* ws = (char*)d_ws;
  auto take = [&](size_t bytes) { char* p = ws; ws += align256(bytes); return p; };
  float*    p_rep   = (float*)take((size_t)B_ * EMB_ * 4);
  float*    h_rep   = (float*)take((size_t)B_ * EMB_ * 4);
  float*    word_p  = (float*)take((size_t)M_nodes * EMB_ * 4);
  float*    word_h  = (float*)take((size_t)M_nodes * EMB_ * 4);
  _Float16* Hp16    = (_Float16*)take((size_t)M_nodes * LDE_ * 2);
  _Float16* Hh16    = (_Float16*)take((size_t)M_nodes * LDE_ * 2);
  _Float16* T16     = (_Float16*)take((size_t)3 * M_nodes * LDE_ * 2);
  _Float16* x16     = (_Float16*)take((size_t)B_ * LDX_ * 2);
  _Float16* h1_16   = (_Float16*)take((size_t)B_ * LDH_ * 2);
  _Float16* h2_16   = (_Float16*)take((size_t)B_ * LDH_ * 2);
  _Float16* h3_16   = (_Float16*)take((size_t)B_ * LDH_ * 2);
  _Float16* W_enc16 = (_Float16*)take((size_t)3 * LDE_ * LDE_ * 2);
  _Float16* W0_16   = (_Float16*)take((size_t)LDH_ * LDX_ * 2);
  _Float16* W1_16   = (_Float16*)take((size_t)LDH_ * LDH_ * 2);
  _Float16* W2_16   = (_Float16*)take((size_t)LDH_ * LDH_ * 2);
  float*    b0p     = (float*)take((size_t)LDH_ * 4);
  float*    b1p     = (float*)take((size_t)LDH_ * 4);
  float*    b2p     = (float*)take((size_t)LDH_ * 4);

  // ---- 0) weight staging: f32 -> zero-padded f16 (tiny, once per launch) ----
  cvt_pad_f16_kernel<<<256, 256, 0, stream>>>(W_enc, W_enc16, EMB_, EMB_, LDE_, LDE_, 3);
  cvt_pad_f16_kernel<<<512, 256, 0, stream>>>(W0_w, W0_16, HID_, 6 * EMB_, LDH_, LDX_, 1);
  cvt_pad_f16_kernel<<<256, 256, 0, stream>>>(W1_w, W1_16, HID_, HID_, LDH_, LDH_, 1);
  cvt_pad_f16_kernel<<<256, 256, 0, stream>>>(W2_w, W2_16, HID_, HID_, LDH_, LDH_, 1);
  pad_bias_kernel<<<(LDH_ + 255) / 256, 256, 0, stream>>>(W0_b, b0p, HID_, LDH_);
  pad_bias_kernel<<<(LDH_ + 255) / 256, 256, 0, stream>>>(W1_b, b1p, HID_, LDH_);
  pad_bias_kernel<<<(LDH_ + 255) / 256, 256, 0, stream>>>(W2_b, b2p, HID_, LDH_);

  // ---- 1) bag-of-words reps + tree leaf/word init (gather, bandwidth bound) ----
  embed_sum_kernel<<<B_, 320, 0, stream>>>(E_tab, premise_x, p_rep);
  embed_sum_kernel<<<B_, 320, 0, stream>>>(E_tab, hypothesis_x, h_rep);
  gather_word_init_kernel<<<dim3(B_, N_), 320, 0, stream>>>(E_tab, premise_x, p_word_idx, word_p, Hp16);
  gather_word_init_kernel<<<dim3(B_, N_), 320, 0, stream>>>(E_tab, hypothesis_x, h_word_idx, word_h, Hh16);

  // ---- 2) tree encoder: per level one batched WMMA GEMM (3 relations) + combine ----
  const dim3 tgrid(M_nodes / 128, LDE_ / 32, 3);
  _Float16*  Hs[2]   = {Hp16, Hh16};
  float*     Wds[2]  = {word_p, word_h};
  const int* rels[2] = {p_rel_id, h_rel_id};
  for (int s = 0; s < 2; ++s) {
    for (int l = 3; l >= 1; --l) {
      gemm_wmma_f16<<<tgrid, 128, 0, stream>>>(
          Hs[s], LDE_, W_enc16, LDE_, (long long)LDE_ * LDE_,
          T16, LDE_, (long long)M_nodes * LDE_,
          LDE_, nullptr, 0);
      tree_combine_kernel<<<dim3(B_, N_), 320, 0, stream>>>(
          Hs[s], Wds[s], T16, rels[s], parent, level, l);
    }
  }

  // ---- 3) feature concat + MLP (branch-free WMMA, fused bias+ReLU) ----
  build_x_kernel<<<B_, 320, 0, stream>>>(p_rep, h_rep, Hp16, Hh16, x16);
  const dim3 mgrid(B_ / 128, LDH_ / 32, 1);
  gemm_wmma_f16<<<mgrid, 128, 0, stream>>>(x16, LDX_, W0_16, LDX_, 0,
                                           h1_16, LDH_, 0, LDX_, b0p, 1);
  gemm_wmma_f16<<<mgrid, 128, 0, stream>>>(h1_16, LDH_, W1_16, LDH_, 0,
                                           h2_16, LDH_, 0, LDH_, b1p, 1);
  gemm_wmma_f16<<<mgrid, 128, 0, stream>>>(h2_16, LDH_, W2_16, LDH_, 0,
                                           h3_16, LDH_, 0, LDH_, b2p, 1);

  // ---- 4) tiny 600->3 output layer ----
  out_layer_kernel<<<B_, 96, 0, stream>>>(h3_16, Wout_w, Wout_b, out);
}
